// EFGNN_10075993276497
// MI455X (gfx1250) — matrix operand order repaired and checked by reading, not compiled, not run on hardware
//
#include <hip/hip_runtime.h>
#include <math.h>

// ---------------------------------------------------------------------------
// EFGNN forward for MI455X (gfx1250, wave32, WMMA) — round 3
// ---------------------------------------------------------------------------

typedef __attribute__((ext_vector_type(16))) _Float16 v16h;
typedef __attribute__((ext_vector_type(8)))  float    v8f;

union Frag16 { v16h v; unsigned u[8]; uint4 q[2]; };

#define N_NODES 50000
#define N_EDGES 800000
#define N_REDGES 850000
#define IN_DIM 500
#define H4 512
#define H2 256
#define HID 128
#define OUTD 40

// ---------------------------------------------------------------------------
// prep kernels
// ---------------------------------------------------------------------------
__global__ void prep_a_kernel(const float* __restrict__ alpha,
                              const float* __restrict__ gamma,
                              float* __restrict__ aVec) {
  if (threadIdx.x == 0 && blockIdx.x == 0) {
    float mx = -1e30f;
    for (int i = 0; i < 7; ++i) mx = fmaxf(mx, alpha[i]);
    float e[7], s = 0.f;
    for (int i = 0; i < 7; ++i) { e[i] = expf(alpha[i] - mx); s += e[i]; }
    float g = gamma[0];
    for (int i = 0; i < 7; ++i) aVec[i] = g * e[i] / s;
  }
}

__global__ void prep_scales_kernel(const float* __restrict__ h1,
                                   const float* __restrict__ renorm,
                                   float* __restrict__ h1inv,
                                   float* __restrict__ h1isqrt,
                                   float* __restrict__ rinv, int n) {
  int i = blockIdx.x * blockDim.x + threadIdx.x;
  int stride = gridDim.x * blockDim.x;
  for (; i < n; i += stride) {
    float h = h1[i];
    h1inv[i] = 1.0f / h;
    h1isqrt[i] = rsqrtf(h);
    rinv[i] = 1.0f / renorm[i];
  }
}

__global__ void zero4_kernel(float4* __restrict__ p, long long n4) {
  long long i = (long long)blockIdx.x * blockDim.x + threadIdx.x;
  long long stride = (long long)gridDim.x * blockDim.x;
  float4 z = make_float4(0.f, 0.f, 0.f, 0.f);
  for (; i < n4; i += stride) p[i] = z;
}

// f32 [gridDim.y][K] -> f16 [gridDim.y][Kp], zero-padded K..Kp
__global__ void cvt_pad_kernel(const float* __restrict__ src,
                               _Float16* __restrict__ dst, int K, int Kp) {
  int m = blockIdx.y;
  int k = blockIdx.x * blockDim.x + threadIdx.x;
  if (k >= Kp) return;
  float v = (k < K) ? src[(long long)m * K + k] : 0.f;
  dst[(long long)m * Kp + k] = (_Float16)v;
}

// f32 src[Ksrc][Nsrc] -> f16 dst[gridDim.y][Kp] transposed, zero-padded
__global__ void transpose_cvt_kernel(const float* __restrict__ src,
                                     _Float16* __restrict__ dst,
                                     int Ksrc, int Nsrc, int Kp) {
  int n = blockIdx.y;
  int k = blockIdx.x * blockDim.x + threadIdx.x;
  if (k >= Kp) return;
  float v = (k < Ksrc && n < Nsrc) ? src[(long long)k * Nsrc + n] : 0.f;
  dst[(long long)n * Kp + k] = (_Float16)v;
}

// ---------------------------------------------------------------------------
// WMMA GEMM: C[M,N] = A[M,Kp] @ Bt[N,Kp]^T + bias[N]
// A,Bt pre-converted f16 (zero-padded K). block 256 = 8 waves (4M x 2N),
// block tile 128x128, wave tile 32x64 (2x4 WMMA tiles): 8 wmma per 12
// ds_load_b128 per K-step. Branch-free b128 staging.
// ---------------------------------------------------------------------------
__global__ __launch_bounds__(256) void gemm_wmma_kernel(
    const _Float16* __restrict__ A, const _Float16* __restrict__ Bt,
    const float* __restrict__ bias, float* __restrict__ C,
    int M, int Kp, int N) {
  __shared__ __align__(16) _Float16 Als[128 * 32];  // [m][k]
  __shared__ __align__(16) _Float16 Bls[128 * 32];  // [n][k]
  const int tid = threadIdx.x;
  const int wave = tid >> 5, lane = tid & 31;
  const int wm = wave >> 1, wn = wave & 1;          // 4(M) x 2(N)
  const int bm = blockIdx.x * 128, bn = blockIdx.y * 128;
  const int ml = lane & 15, hi = lane >> 4;

  const int sr = tid >> 2;          // staging row (0..63)
  const int sc = (tid & 3) * 8;     // staging chunk col (halves)

  v8f acc[2][4] = {};
  for (int kt = 0; kt < Kp; kt += 32) {
    // A tile: 128 rows x 32 halves, two b128 chunks per thread
    #pragma unroll
    for (int i = 0; i < 2; ++i) {
      int m = sr + 64 * i;
      int gm = bm + m; if (gm > M - 1) gm = M - 1;   // clamp: branch-free
      *(uint4*)&Als[m * 32 + sc] =
          *(const uint4*)(A + (long long)gm * Kp + kt + sc);
    }
    // B tile: 128 rows x 32 halves, two b128 chunks per thread
    #pragma unroll
    for (int i = 0; i < 2; ++i) {
      int n = sr + 64 * i;
      *(uint4*)&Bls[n * 32 + sc] =
          *(const uint4*)(Bt + (long long)(bn + n) * Kp + kt + sc);
    }
    __syncthreads();

    Frag16 af[2], bf[4];
    #pragma unroll
    for (int t = 0; t < 2; ++t) {
      int mrow = wm * 32 + t * 16 + ml;
      af[t].q[0] = *(const uint4*)&Als[mrow * 32 + 8 * hi];
      af[t].q[1] = *(const uint4*)&Als[mrow * 32 + 16 + 8 * hi];
    }
    #pragma unroll
    for (int t = 0; t < 4; ++t) {
      int ncol = wn * 64 + t * 16 + ml;
      bf[t].q[0] = *(const uint4*)&Bls[ncol * 32 + 8 * hi];
      bf[t].q[1] = *(const uint4*)&Bls[ncol * 32 + 16 + 8 * hi];
    }
    #pragma unroll
    for (int tm = 0; tm < 2; ++tm)
      #pragma unroll
      for (int tn = 0; tn < 4; ++tn)
        acc[tm][tn] = __builtin_amdgcn_wmma_f32_16x16x32_f16(
            false, af[tm].v, false, bf[tn].v, (short)0, acc[tm][tn], false, false);
    __syncthreads();
  }

  #pragma unroll
  for (int tm = 0; tm < 2; ++tm)
    #pragma unroll
    for (int tn = 0; tn < 4; ++tn) {
      int col = bn + wn * 64 + tn * 16 + ml;
      float bb = bias[col];
      #pragma unroll
      for (int r = 0; r < 8; ++r) {
        int row = bm + wm * 32 + tm * 16 + r + 8 * hi;
        if (row < M) C[(long long)row * N + col] = acc[tm][tn][r] + bb;
      }
    }
}

// ---------------------------------------------------------------------------
// SpMM scatter: out[row[e], :] += w_e * scale[col[e]] * in[col[e], :]
// one wave32 per edge; 128 features = 32 lanes x float4; f32 atomics (L2-hot)
// ---------------------------------------------------------------------------
__global__ void spmm_scatter_kernel(
    float* __restrict__ out,                 // ld = 128
    const float* __restrict__ in, int ldin,
    const int* __restrict__ rows, const int* __restrict__ cols,
    const float* __restrict__ eval,          // nullable edge values
    const float* __restrict__ nscale,        // nullable per-src-node scale
    int nE) {
  int wave = (blockIdx.x * blockDim.x + threadIdx.x) >> 5;
  int lane = threadIdx.x & 31;
  int nW = (gridDim.x * blockDim.x) >> 5;
  for (int e = wave; e < nE; e += nW) {
    if (e + nW < nE) __builtin_prefetch(&cols[e + nW], 0, 0);
    int r = rows[e], c = cols[e];
    float w = eval ? eval[e] : 1.0f;
    if (nscale) w *= nscale[c];
    const float4 v = *(const float4*)(in + (long long)c * ldin + lane * 4);
    float* dst = out + (long long)r * HID + lane * 4;
    atomicAdd(dst + 0, w * v.x);
    atomicAdd(dst + 1, w * v.y);
    atomicAdd(dst + 2, w * v.z);
    atomicAdd(dst + 3, w * v.w);
  }
}

// ---------------------------------------------------------------------------
// combine + l2norm + leaky -> f16:
//   dst = (f16) leaky(l2norm(k1*s1[i]*d1 + k2*s2[i]*d2))
// one wave per row (128 feats = 32 lanes x float4)
// ---------------------------------------------------------------------------
__global__ void combine_norm_leaky_kernel(
    _Float16* __restrict__ dst, int lddst,
    const float* __restrict__ d1, int ld1, const float* __restrict__ s1, float k1,
    const float* __restrict__ d2, int ld2, const float* __restrict__ s2, float k2,
    int M) {
  int wave = (blockIdx.x * blockDim.x + threadIdx.x) >> 5;
  int lane = threadIdx.x & 31;
  int nW = (gridDim.x * blockDim.x) >> 5;
  for (int i = wave; i < M; i += nW) {
    float w1 = k1 * (s1 ? s1[i] : 1.0f);
    float4 v = *(const float4*)(d1 + (long long)i * ld1 + lane * 4);
    float zx = w1 * v.x, zy = w1 * v.y, zz = w1 * v.z, zw = w1 * v.w;
    if (d2) {
      float w2 = k2 * (s2 ? s2[i] : 1.0f);
      float4 u = *(const float4*)(d2 + (long long)i * ld2 + lane * 4);
      zx += w2 * u.x; zy += w2 * u.y; zz += w2 * u.z; zw += w2 * u.w;
    }
    float ss = zx * zx + zy * zy + zz * zz + zw * zw;
    ss += __shfl_xor(ss, 1, 32);
    ss += __shfl_xor(ss, 2, 32);
    ss += __shfl_xor(ss, 4, 32);
    ss += __shfl_xor(ss, 8, 32);
    ss += __shfl_xor(ss, 16, 32);
    float sc = 1.0f / fmaxf(sqrtf(ss), 1e-12f);
    zx *= sc; zy *= sc; zz *= sc; zw *= sc;
    zx = zx >= 0.f ? zx : 0.01f * zx;
    zy = zy >= 0.f ? zy : 0.01f * zy;
    zz = zz >= 0.f ? zz : 0.01f * zz;
    zw = zw >= 0.f ? zw : 0.01f * zw;
    union { _Float16 h[4]; uint2 u; } o;
    o.h[0] = (_Float16)zx; o.h[1] = (_Float16)zy;
    o.h[2] = (_Float16)zz; o.h[3] = (_Float16)zw;
    *(uint2*)(dst + (long long)i * lddst + lane * 4) = o.u;
  }
}

// ---------------------------------------------------------------------------
// out projection: out[M,40] += a * l2norm_row(Z[M,128] @ W[128,40] + b[40])
// Z is f16 (ldz), Wt is f16 pre-transposed [48][128]. block 256 = 8 waves,
// each wave 16 rows; 3 WMMA n-tiles (40 padded to 48).
// ---------------------------------------------------------------------------
__global__ __launch_bounds__(256) void out_proj_wmma_kernel(
    const _Float16* __restrict__ Z, int ldz,
    const _Float16* __restrict__ Wt,   // [48][128] f16, n-padded
    const float* __restrict__ bout,    // 40
    const float* __restrict__ aVec, int aIdx,
    float* __restrict__ out, int M) {
  __shared__ __align__(16) _Float16 Als[128 * 128]; // [m][k]
  __shared__ __align__(16) _Float16 Bls[48 * 128];  // [n][k]
  const int tid = threadIdx.x, wave = tid >> 5, lane = tid & 31;
  const int bm = blockIdx.x * 128;
  const int ml = lane & 15, hi = lane >> 4;

  // stage Z rows: 128 x 128 halves = 2048 b128 chunks, 8 per thread
  #pragma unroll
  for (int i = 0; i < 8; ++i) {
    int id = tid + 256 * i;
    int m = id >> 4, c = (id & 15) * 8;
    int gm = bm + m; if (gm > M - 1) gm = M - 1;
    *(uint4*)&Als[m * 128 + c] = *(const uint4*)(Z + (long long)gm * ldz + c);
  }
  // stage Wt: 48 x 128 halves = 768 chunks, 3 per thread
  #pragma unroll
  for (int i = 0; i < 3; ++i) {
    int id = tid + 256 * i;
    int n = id >> 4, c = (id & 15) * 8;
    *(uint4*)&Bls[n * 128 + c] = *(const uint4*)(Wt + n * 128 + c);
  }
  __syncthreads();

  v8f acc[3] = {};
  #pragma unroll
  for (int kt = 0; kt < 128; kt += 32) {
    Frag16 a;
    int mrow = wave * 16 + ml;
    a.q[0] = *(const uint4*)&Als[mrow * 128 + kt + 8 * hi];
    a.q[1] = *(const uint4*)&Als[mrow * 128 + kt + 16 + 8 * hi];
    #pragma unroll
    for (int tn = 0; tn < 3; ++tn) {
      Frag16 b;
      int ncol = tn * 16 + ml;
      b.q[0] = *(const uint4*)&Bls[ncol * 128 + kt + 8 * hi];
      b.q[1] = *(const uint4*)&Bls[ncol * 128 + kt + 16 + 8 * hi];
      acc[tn] = __builtin_amdgcn_wmma_f32_16x16x32_f16(
          false, a.v, false, b.v, (short)0, acc[tn], false, false);
    }
  }

  // epilogue: bias, per-row l2 norm (cross-lane reduce), scaled accumulate
  const float av = aVec[aIdx];
  float vals[3][8];
  float ss[8];
  #pragma unroll
  for (int r = 0; r < 8; ++r) ss[r] = 0.f;
  #pragma unroll
  for (int tn = 0; tn < 3; ++tn) {
    int col = tn * 16 + ml;
    float bb = (col < OUTD) ? bout[col] : 0.f;
    #pragma unroll
    for (int r = 0; r < 8; ++r) {
      float v = acc[tn][r] + bb;
      if (col >= OUTD) v = 0.f;
      vals[tn][r] = v;
      ss[r] += v * v;
    }
  }
  #pragma unroll
  for (int r = 0; r < 8; ++r) {
    float s = ss[r];
    s += __shfl_xor(s, 1, 32);
    s += __shfl_xor(s, 2, 32);
    s += __shfl_xor(s, 4, 32);
    s += __shfl_xor(s, 8, 32);
    float sc = av / fmaxf(sqrtf(s), 1e-12f);
    int row = bm + wave * 16 + r + 8 * hi;
    if (row < M) {
      #pragma unroll
      for (int tn = 0; tn < 3; ++tn) {
        int col = tn * 16 + ml;
        if (col < OUTD)
          out[(long long)row * OUTD + col] += sc * vals[tn][r];
      }
    }
  }
}

// ---------------------------------------------------------------------------
// final: out[i,:] += a6*(h1_deg[i]*W_deg + b_deg); log_softmax rows
// ---------------------------------------------------------------------------
__global__ void final_logsoftmax_kernel(
    float* __restrict__ out, const float* __restrict__ h1deg,
    const float* __restrict__ Wdeg, const float* __restrict__ bdeg,
    const float* __restrict__ aVec, int M) {
  int i = blockIdx.x * blockDim.x + threadIdx.x;
  if (i >= M) return;
  float a6 = aVec[6];
  float d = h1deg[i];
  float v[OUTD];
  float mx = -1e30f;
  #pragma unroll
  for (int j = 0; j < OUTD; ++j) {
    float t = out[(long long)i * OUTD + j] + a6 * (d * Wdeg[j] + bdeg[j]);
    v[j] = t;
    mx = fmaxf(mx, t);
  }
  float s = 0.f;
  #pragma unroll
  for (int j = 0; j < OUTD; ++j) s += expf(v[j] - mx);
  float lse = mx + logf(s);
  #pragma unroll
  for (int j = 0; j < OUTD; ++j) out[(long long)i * OUTD + j] = v[j] - lse;
}

// ---------------------------------------------------------------------------
// launch
// ---------------------------------------------------------------------------
extern "C" void kernel_launch(void* const* d_in, const int* in_sizes, int n_in,
                              void* d_out, int out_size, void* d_ws, size_t ws_size,
                              hipStream_t stream) {
  (void)in_sizes; (void)n_in; (void)out_size; (void)ws_size;
  const float* x          = (const float*)d_in[0];
  const float* h1_deg     = (const float*)d_in[1];
  const float* renorm_deg = (const float*)d_in[2];
  const float* r_val      = (const float*)d_in[3];
  const float* W1         = (const float*)d_in[4];
  const float* b1         = (const float*)d_in[5];
  const float* W_dp       = (const float*)d_in[6];
  const float* b_dp       = (const float*)d_in[7];
  const float* W_out      = (const float*)d_in[8];   // 6 x 128 x 40
  const float* b_out      = (const float*)d_in[9];   // 6 x 40
  const float* W_deg      = (const float*)d_in[10];  // 1 x 40
  const float* b_deg      = (const float*)d_in[11];
  const float* alpha      = (const float*)d_in[12];
  const float* gamma      = (const float*)d_in[13];
  const int* edge_row     = (const int*)d_in[14];
  const int* edge_col     = (const int*)d_in[15];
  const int* r_row        = (const int*)d_in[16];
  const int* r_col        = (const int*)d_in[17];
  float* out = (float*)d_out;

  char* wsb = (char*)d_ws;
  size_t off = 0;
  auto alloc = [&](size_t bytes) -> void* {
    void* p = wsb + off;
    off += (bytes + 255) & ~(size_t)255;
    return p;
  };
  float* aVec    = (float*)alloc(8 * 4);
  float* rinv    = (float*)alloc((size_t)N_NODES * 4);
  float* h1inv   = (float*)alloc((size_t)N_NODES * 4);
  float* h1isqrt = (float*)alloc((size_t)N_NODES * 4);
  _Float16* Xh   = (_Float16*)alloc((size_t)N_NODES * H4 * 2);
  _Float16* W1t  = (_Float16*)alloc((size_t)H4 * H4 * 2);      // [512][512]
  _Float16* WDPt = (_Float16*)alloc((size_t)H2 * H4 * 2);      // [256][512]
  _Float16* WOt  = (_Float16*)alloc((size_t)6 * 48 * HID * 2); // 6 x [48][128]
  float* X1      = (float*)alloc((size_t)N_NODES * H4 * 4);
  _Float16* ZSh  = (_Float16*)alloc((size_t)N_NODES * H4 * 2);
  float* DP      = (float*)alloc((size_t)N_NODES * H2 * 4);
  float* B1      = (float*)alloc((size_t)N_NODES * HID * 4);
  float* B2      = (float*)alloc((size_t)N_NODES * HID * 4);
  _Float16* B2h  = (_Float16*)alloc((size_t)N_NODES * HID * 2);

  const int GM = (N_NODES + 127) / 128;   // 391
  const long long nb4 = (long long)N_NODES * HID / 4;
  const int SPB = 8192;
  const int CNB = 6250;   // 50000 waves

  // ---- prep: scalars, scales, conversions ----
  prep_a_kernel<<<1, 32, 0, stream>>>(alpha, gamma, aVec);
  prep_scales_kernel<<<196, 256, 0, stream>>>(h1_deg, renorm_deg, h1inv, h1isqrt, rinv, N_NODES);
  zero4_kernel<<<2048, 256, 0, stream>>>((float4*)out, (long long)N_NODES * OUTD / 4);

  cvt_pad_kernel<<<dim3(2, N_NODES), 256, 0, stream>>>(x, Xh, IN_DIM, H4);
  transpose_cvt_kernel<<<dim3(2, H4), 256, 0, stream>>>(W1, W1t, IN_DIM, H4, H4);
  transpose_cvt_kernel<<<dim3(2, H2), 256, 0, stream>>>(W_dp, WDPt, H4, H2, H4);
  for (int i = 0; i < 6; ++i)
    transpose_cvt_kernel<<<dim3(1, 48), 256, 0, stream>>>(
        W_out + (size_t)i * HID * OUTD, WOt + (size_t)i * 48 * HID, HID, OUTD, HID);

  // x1 = x @ W1 + b1   (block tile 128x128)
  gemm_wmma_kernel<<<dim3(GM, H4 / 128), 256, 0, stream>>>(Xh, W1t, b1, X1, N_NODES, H4, H4);

  // --- Filter 0: res = rinv * spmmr(rinv*z0) ---
  zero4_kernel<<<2048, 256, 0, stream>>>((float4*)B1, nb4);
  spmm_scatter_kernel<<<SPB, 256, 0, stream>>>(B1, X1 + 0, H4, r_row, r_col, r_val, rinv, N_REDGES);
  combine_norm_leaky_kernel<<<CNB, 256, 0, stream>>>(ZSh + 0, H4, B1, HID, rinv, 1.0f,
                                                     nullptr, 0, nullptr, 0.0f, N_NODES);
  out_proj_wmma_kernel<<<GM, 256, 0, stream>>>(ZSh + 0, H4, WOt + 0 * 48 * HID,
                                               b_out + 0 * OUTD, aVec, 0, out, N_NODES);

  // --- Filter 1: res = spmm1(h1inv*z1) ---
  zero4_kernel<<<2048, 256, 0, stream>>>((float4*)B1, nb4);
  spmm_scatter_kernel<<<SPB, 256, 0, stream>>>(B1, X1 + HID, H4, edge_row, edge_col, nullptr, h1inv, N_EDGES);
  combine_norm_leaky_kernel<<<CNB, 256, 0, stream>>>(ZSh + HID, H4, B1, HID, nullptr, 1.0f,
                                                     nullptr, 0, nullptr, 0.0f, N_NODES);
  out_proj_wmma_kernel<<<GM, 256, 0, stream>>>(ZSh + HID, H4, WOt + 1 * 48 * HID,
                                               b_out + 1 * OUTD, aVec, 1, out, N_NODES);

  // --- Filter 2: res = rinv * spmmr(spmmr(rinv*z2)) ---
  zero4_kernel<<<2048, 256, 0, stream>>>((float4*)B1, nb4);
  spmm_scatter_kernel<<<SPB, 256, 0, stream>>>(B1, X1 + 2 * HID, H4, r_row, r_col, r_val, rinv, N_REDGES);
  zero4_kernel<<<2048, 256, 0, stream>>>((float4*)B2, nb4);
  spmm_scatter_kernel<<<SPB, 256, 0, stream>>>(B2, B1, HID, r_row, r_col, r_val, nullptr, N_REDGES);
  combine_norm_leaky_kernel<<<CNB, 256, 0, stream>>>(ZSh + 2 * HID, H4, B2, HID, rinv, 1.0f,
                                                     nullptr, 0, nullptr, 0.0f, N_NODES);
  out_proj_wmma_kernel<<<GM, 256, 0, stream>>>(ZSh + 2 * HID, H4, WOt + 2 * 48 * HID,
                                               b_out + 2 * OUTD, aVec, 2, out, N_NODES);

  // --- Filter 3: res = -rinv*spmmr(rinv*z3) + h1inv*spmm1(z3) ---
  zero4_kernel<<<2048, 256, 0, stream>>>((float4*)B1, nb4);
  spmm_scatter_kernel<<<SPB, 256, 0, stream>>>(B1, X1 + 3 * HID, H4, r_row, r_col, r_val, rinv, N_REDGES);
  zero4_kernel<<<2048, 256, 0, stream>>>((float4*)B2, nb4);
  spmm_scatter_kernel<<<SPB, 256, 0, stream>>>(B2, X1 + 3 * HID, H4, edge_row, edge_col, nullptr, nullptr, N_EDGES);
  combine_norm_leaky_kernel<<<CNB, 256, 0, stream>>>(ZSh + 3 * HID, H4, B1, HID, rinv, -1.0f,
                                                     B2, HID, h1inv, 1.0f, N_NODES);
  out_proj_wmma_kernel<<<GM, 256, 0, stream>>>(ZSh + 3 * HID, H4, WOt + 3 * 48 * HID,
                                               b_out + 3 * OUTD, aVec, 3, out, N_NODES);

  // dp = x_cat @ W_dp + b_dp  (x_cat == ZSh, already f16)
  gemm_wmma_kernel<<<dim3(GM, H2 / 128), 256, 0, stream>>>(ZSh, WDPt, b_dp, DP, N_NODES, H4, H2);

  // --- PROP2[0]: z = h1^-.5 * spmm1(h1^-.5 * ego) + ego ---
  zero4_kernel<<<2048, 256, 0, stream>>>((float4*)B1, nb4);
  spmm_scatter_kernel<<<SPB, 256, 0, stream>>>(B1, DP + 0, H2, edge_row, edge_col, nullptr, h1isqrt, N_EDGES);
  combine_norm_leaky_kernel<<<CNB, 256, 0, stream>>>(B2h, HID, B1, HID, h1isqrt, 1.0f,
                                                     DP + 0, H2, nullptr, 1.0f, N_NODES);
  out_proj_wmma_kernel<<<GM, 256, 0, stream>>>(B2h, HID, WOt + 0 * 48 * HID,
                                               b_out + 0 * OUTD, aVec, 4, out, N_NODES);

  // --- PROP2[1]: z = spmm1(h1^-1 * ego) ---
  zero4_kernel<<<2048, 256, 0, stream>>>((float4*)B1, nb4);
  spmm_scatter_kernel<<<SPB, 256, 0, stream>>>(B1, DP + HID, H2, edge_row, edge_col, nullptr, h1inv, N_EDGES);
  combine_norm_leaky_kernel<<<CNB, 256, 0, stream>>>(B2h, HID, B1, HID, nullptr, 1.0f,
                                                     nullptr, 0, nullptr, 0.0f, N_NODES);
  out_proj_wmma_kernel<<<GM, 256, 0, stream>>>(B2h, HID, WOt + 1 * 48 * HID,
                                               b_out + 1 * OUTD, aVec, 5, out, N_NODES);

  // final degree term + log_softmax
  final_logsoftmax_kernel<<<(N_NODES + 255) / 256, 256, 0, stream>>>(
      out, h1_deg, W_deg, b_deg, aVec, N_NODES);
}
